// Decoder_70909910057495
// MI455X (gfx1250) — compile-verified, matrix operand order checked by reading, not run on hardware
//
#include <hip/hip_runtime.h>
#include <hip/hip_bf16.h>

typedef __attribute__((ext_vector_type(16))) __bf16 v16bf;
typedef __attribute__((ext_vector_type(8)))  float  v8f;

#define BATCH 2048
#define SEQ   512
#define H2E   64      // 2*ENC
#define DECH  32
#define EMBH  16
#define SCH   8       // sequence chunks
#define SPC   (SEQ / SCH)        // 64 s-positions per wave
#define TILES (BATCH / 16)       // 128 batch tiles

// ---- branch-free transcendentals on raw v_exp_f32 / v_rcp_f32 -------------
__device__ __forceinline__ float fast_tanh(float x) {
    // tanh(x) = 1 - 2/(exp(2x)+1); exp2-based, saturates to +/-1, no branches
    float t = __builtin_amdgcn_exp2f(x * 2.8853900817779268f);   // 2*log2(e)
    float r = __builtin_amdgcn_rcpf(t + 1.0f);
    return fmaf(-2.0f, r, 1.0f);
}
__device__ __forceinline__ float fast_exp(float x) {
    return __builtin_amdgcn_exp2f(x * 1.4426950408889634f);      // log2(e)
}
__device__ __forceinline__ float fast_sigmoid(float x) {
    float t = __builtin_amdgcn_exp2f(-x * 1.4426950408889634f);
    return __builtin_amdgcn_rcpf(1.0f + t);
}

// ---------------------------------------------------------------------------
// Kernel 1: fused attention pass. Each wave32 owns a 16-row batch tile and a
// 64-long sequence chunk. Streams enc_out once, WMMA energy, online exp-sum
// and context accumulation in registers. Writes partial (sum w, sum w*enc).
// ---------------------------------------------------------------------------
__global__ __launch_bounds__(128)
void attn_fused_kernel(const float* __restrict__ enc,    // (S,B,64)
                       const float* __restrict__ sh,     // (B,32)
                       const float* __restrict__ Wattn,  // (32,96)
                       const float* __restrict__ battn,  // (32)
                       const float* __restrict__ vattn,  // (32)
                       float* __restrict__ c_part,       // (SCH,B,64)
                       float* __restrict__ l_part)       // (SCH,B)
{
    const int lane = threadIdx.x & 31;
    const int wid  = blockIdx.x * 4 + (threadIdx.x >> 5);   // 0..1023
    const int tile = wid & (TILES - 1);
    const int sch  = wid >> 7;
    const int b0   = tile * 16;
    const int m    = lane & 15;          // row within tile
    const int hib  = (lane >> 4) * 8;    // 0 for lanes 0-15, 8 for lanes 16-31

    // ---- preload B fragments of W_e (bf16), K=64 split in 2, N=32 in 2 ----
    v16bf Bf[2][2];
#pragma unroll
    for (int c = 0; c < 2; ++c)
#pragma unroll
        for (int t = 0; t < 2; ++t) {
#pragma unroll
            for (int i = 0; i < 16; ++i) {
                int K  = (i < 8 ? i : i + 8) + hib;          // per-lane K slot
                int kg = c * 32 + K;                          // global k in 0..63
                int n  = t * 16 + m;                          // output channel
                Bf[c][t][i] = (__bf16)Wattn[n * 96 + 32 + kg];
            }
        }

    // ---- per-tile s-bias: sb[row][n] = b_attn[n] + W_s @ s[b0+row] --------
    float sb0[8], sb1[8];
    {
        const int n0 = m, n1 = 16 + m;
#pragma unroll
        for (int j = 0; j < 8; ++j) {
            int row = j + hib;
            float a0 = battn[n0], a1 = battn[n1];
            const float* sr = sh + (b0 + row) * DECH;
#pragma unroll 8
            for (int k = 0; k < DECH; ++k) {
                float sv = sr[k];
                a0 = fmaf(sv, Wattn[n0 * 96 + k], a0);
                a1 = fmaf(sv, Wattn[n1 * 96 + k], a1);
            }
            sb0[j] = a0; sb1[j] = a1;
        }
    }
    const float vv0 = vattn[m], vv1 = vattn[16 + m];

    float cacc[32];
#pragma unroll
    for (int k = 0; k < 32; ++k) cacc[k] = 0.0f;
    float lsum = 0.0f;

    const float* rowbase =
        enc + ((size_t)(sch * SPC) * BATCH + b0 + m) * H2E;

    for (int si = 0; si < SPC; ++si) {
        const float* p = rowbase + (size_t)si * (BATCH * H2E);
        // lane's 32 f32 values of the 16x64 enc tile (2 lanes cover one row)
        float fr[32];
#pragma unroll
        for (int g = 0; g < 4; ++g) {
            int off = g * 16 + hib;
            float4 u0 = *(const float4*)(p + off);
            float4 u1 = *(const float4*)(p + off + 4);
            fr[g*8+0]=u0.x; fr[g*8+1]=u0.y; fr[g*8+2]=u0.z; fr[g*8+3]=u0.w;
            fr[g*8+4]=u1.x; fr[g*8+5]=u1.y; fr[g*8+6]=u1.z; fr[g*8+7]=u1.w;
        }
        __builtin_prefetch(p + (size_t)(BATCH * H2E), 0, 3);

        // A fragments (bf16) for the two K-chunks
        v16bf a0, a1;
#pragma unroll
        for (int i = 0; i < 16; ++i) {
            a0[i] = (__bf16)fr[i];
            a1[i] = (__bf16)fr[16 + i];
        }

        // energy = tanh(Enc @ W_e^T + sbias); C preloaded with s-bias.
        // Interleave acc0/acc1 chains to hide WMMA->WMMA RAW hazard NOPs.
        v8f acc0, acc1;
#pragma unroll
        for (int j = 0; j < 8; ++j) { acc0[j] = sb0[j]; acc1[j] = sb1[j]; }
        acc0 = __builtin_amdgcn_wmma_f32_16x16x32_bf16(false, a0, false, Bf[0][0], (short)0, acc0, false, false);
        acc1 = __builtin_amdgcn_wmma_f32_16x16x32_bf16(false, a0, false, Bf[0][1], (short)0, acc1, false, false);
        acc0 = __builtin_amdgcn_wmma_f32_16x16x32_bf16(false, a1, false, Bf[1][0], (short)0, acc0, false, false);
        acc1 = __builtin_amdgcn_wmma_f32_16x16x32_bf16(false, a1, false, Bf[1][1], (short)0, acc1, false, false);

        // per-lane contribution of column N=m to each of its 8 rows
        float pj[8];
#pragma unroll
        for (int j = 0; j < 8; ++j)
            pj[j] = fast_tanh(acc0[j]) * vv0 + fast_tanh(acc1[j]) * vv1;

        // butterfly transpose-reduce over the 16-lane half: after this, lane
        // holds the full logit of row (m&7) + 8*(lane>=16)
        const bool s0 = (m & 1), s1 = (m & 2), s2 = (m & 4);
        float q0 = (s0 ? pj[1] : pj[0]) + __shfl_xor(s0 ? pj[0] : pj[1], 1, 32);
        float q1 = (s0 ? pj[3] : pj[2]) + __shfl_xor(s0 ? pj[2] : pj[3], 1, 32);
        float q2 = (s0 ? pj[5] : pj[4]) + __shfl_xor(s0 ? pj[4] : pj[5], 1, 32);
        float q3 = (s0 ? pj[7] : pj[6]) + __shfl_xor(s0 ? pj[6] : pj[7], 1, 32);
        float r0 = (s1 ? q1 : q0) + __shfl_xor(s1 ? q0 : q1, 2, 32);
        float r1 = (s1 ? q3 : q2) + __shfl_xor(s1 ? q2 : q3, 2, 32);
        float tt = (s2 ? r1 : r0) + __shfl_xor(s2 ? r0 : r1, 4, 32);
        tt += __shfl_xor(tt, 8, 32);

        // gather logit of this lane's own row m from the correct half
        const int src = (m & 7) | ((m & 8) << 1);
        float w = fast_exp(__shfl(tt, src, 32));   // logits tiny: no max needed

        lsum += w;
#pragma unroll
        for (int k = 0; k < 32; ++k) cacc[k] = fmaf(w, fr[k], cacc[k]);
    }

    // write partial results (each slot owned by exactly one wave/lane)
    float* cp = c_part + ((size_t)sch * BATCH + b0 + m) * H2E;
#pragma unroll
    for (int g = 0; g < 4; ++g) {
        int off = g * 16 + hib;
        float4 u0 = { cacc[g*8+0], cacc[g*8+1], cacc[g*8+2], cacc[g*8+3] };
        float4 u1 = { cacc[g*8+4], cacc[g*8+5], cacc[g*8+6], cacc[g*8+7] };
        *(float4*)(cp + off)     = u0;
        *(float4*)(cp + off + 4) = u1;
    }
    if (lane < 16)
        l_part[(size_t)sch * BATCH + b0 + m] = lsum;
}

// ---------------------------------------------------------------------------
// Kernel 2: combine partial sums -> ctx[b][h] = sum(c)/sum(l)
// ---------------------------------------------------------------------------
__global__ void ctx_finalize_kernel(const float* __restrict__ c_part,
                                    const float* __restrict__ l_part,
                                    float* __restrict__ ctx)
{
    int idx = blockIdx.x * blockDim.x + threadIdx.x;
    if (idx >= BATCH * H2E) return;
    int b = idx >> 6;
    float num = 0.0f, den = 0.0f;
#pragma unroll
    for (int k = 0; k < SCH; ++k) {
        num += c_part[(size_t)k * BATCH * H2E + idx];
        den += l_part[k * BATCH + b];
    }
    ctx[idx] = num / den;
}

// ---------------------------------------------------------------------------
// Kernel 3: embedding + GRU step + fc, one thread per batch row (tiny)
// ---------------------------------------------------------------------------
__global__ __launch_bounds__(128)
void decoder_tail_kernel(const float* __restrict__ dec_in,
                         const float* __restrict__ sh,
                         const float* __restrict__ ctx,
                         const float* __restrict__ Wemb, const float* __restrict__ bemb,
                         const float* __restrict__ Wih,  const float* __restrict__ Whh,
                         const float* __restrict__ bih,  const float* __restrict__ bhh,
                         const float* __restrict__ Wfc,  const float* __restrict__ bfc,
                         float* __restrict__ out)
{
    int b = blockIdx.x * blockDim.x + threadIdx.x;
    if (b >= BATCH) return;

    const float x = dec_in[b];
    float xg[EMBH + H2E];                    // [emb(16); ctx(64)] = GRU input
#pragma unroll
    for (int e = 0; e < EMBH; ++e) {
        float v = fmaf(Wemb[e], x, bemb[e]); // W_emb is (16,1)
        xg[e] = v > 0.0f ? v : 0.0f;         // relu
    }
#pragma unroll
    for (int h = 0; h < H2E; ++h) xg[EMBH + h] = ctx[b * H2E + h];

    float pred = bfc[0];
    for (int j = 0; j < DECH; ++j) {         // rolled: indexes only globals
        float gir = bih[j], giz = bih[DECH + j], gin = bih[2 * DECH + j];
#pragma unroll
        for (int k = 0; k < EMBH + H2E; ++k) {
            float xv = xg[k];
            gir = fmaf(Wih[j * 80 + k],              xv, gir);
            giz = fmaf(Wih[(DECH + j) * 80 + k],     xv, giz);
            gin = fmaf(Wih[(2 * DECH + j) * 80 + k], xv, gin);
        }
        float ghr = bhh[j], ghz = bhh[DECH + j], ghn = bhh[2 * DECH + j];
#pragma unroll 8
        for (int k = 0; k < DECH; ++k) {
            float sv = sh[b * DECH + k];
            ghr = fmaf(Whh[j * DECH + k],              sv, ghr);
            ghz = fmaf(Whh[(DECH + j) * DECH + k],     sv, ghz);
            ghn = fmaf(Whh[(2 * DECH + j) * DECH + k], sv, ghn);
        }
        float r = fast_sigmoid(gir + ghr);
        float z = fast_sigmoid(giz + ghz);
        float n = fast_tanh(gin + r * ghn);
        float hj = (1.0f - z) * n + z * sh[b * DECH + j];
        out[BATCH + b * DECH + j] = hj;                 // h_new output
        pred = fmaf(Wfc[j], hj, pred);                  // fc: [h_new; ctx; emb]
    }
#pragma unroll
    for (int h = 0; h < H2E; ++h) pred = fmaf(Wfc[DECH + h], xg[EMBH + h], pred);
#pragma unroll
    for (int e = 0; e < EMBH; ++e) pred = fmaf(Wfc[DECH + H2E + e], xg[e], pred);
    out[b] = pred;                                      // pred output
}

// ---------------------------------------------------------------------------
extern "C" void kernel_launch(void* const* d_in, const int* in_sizes, int n_in,
                              void* d_out, int out_size, void* d_ws, size_t ws_size,
                              hipStream_t stream) {
    const float* dec_in = (const float*)d_in[0];
    const float* s_h    = (const float*)d_in[1];
    const float* enc    = (const float*)d_in[2];
    const float* Wattn  = (const float*)d_in[3];
    const float* battn  = (const float*)d_in[4];
    const float* vattn  = (const float*)d_in[5];
    const float* Wemb   = (const float*)d_in[6];
    const float* bemb   = (const float*)d_in[7];
    const float* Wih    = (const float*)d_in[8];
    const float* Whh    = (const float*)d_in[9];
    const float* bih    = (const float*)d_in[10];
    const float* bhh    = (const float*)d_in[11];
    const float* Wfc    = (const float*)d_in[12];
    const float* bfc    = (const float*)d_in[13];
    float* out = (float*)d_out;

    // workspace layout (floats): c_part | l_part | ctx
    float* ws     = (float*)d_ws;
    float* c_part = ws;                                   // SCH*B*64
    float* l_part = ws + (size_t)SCH * BATCH * H2E;       // SCH*B
    float* ctx    = l_part + (size_t)SCH * BATCH;         // B*64

    // 1024 wave32s: 4 waves per block, 256 blocks
    attn_fused_kernel<<<256, 128, 0, stream>>>(enc, s_h, Wattn, battn, vattn,
                                               c_part, l_part);
    ctx_finalize_kernel<<<(BATCH * H2E + 255) / 256, 256, 0, stream>>>(
        c_part, l_part, ctx);
    decoder_tail_kernel<<<(BATCH + 127) / 128, 128, 0, stream>>>(
        dec_in, s_h, ctx, Wemb, bemb, Wih, Whh, bih, bhh, Wfc, bfc, out);
}
